// SelfAttention_64716567216298
// MI455X (gfx1250) — compile-verified, matrix operand order checked by reading, not run on hardware
//
#include <hip/hip_runtime.h>
#include <cstdint>

#define BATCH  2
#define CCH    64
#define C8V    8
#define NV     4096   // 16*16*16 flattened spatial
#define NSPLIT 2      // flash-decoding split over j
#define JSPAN  (NV / NSPLIT)

typedef __attribute__((ext_vector_type(16))) _Float16     v16h;
typedef __attribute__((ext_vector_type(8)))  float        v8f;
typedef __attribute__((ext_vector_type(4)))  float        v4f;
typedef __attribute__((ext_vector_type(4)))  unsigned int v4u;

union V16H {          // 8 VGPRs viewed as 16 halves or 2x b128
  v16h h;
  v4u  q[2];
};

// ---------------- Phase 1a: Q/K projections (64 -> 8), f16 [b][n][8] ----------------
__global__ __launch_bounds__(256) void qk_proj_kernel(
    const float* __restrict__ x,
    const float* __restrict__ Wq, const float* __restrict__ bq,
    const float* __restrict__ Wk, const float* __restrict__ bk,
    _Float16* __restrict__ qT, _Float16* __restrict__ kT)
{
  __shared__ float sWq[C8V * CCH];
  __shared__ float sWk[C8V * CCH];
  const int tid = threadIdx.x;
  for (int i = tid; i < C8V * CCH; i += 256) { sWq[i] = Wq[i]; sWk[i] = Wk[i]; }
  __syncthreads();

  const int blocksPerB = NV / 256;                       // 16
  const int b = blockIdx.x / blocksPerB;
  const int n = (blockIdx.x % blocksPerB) * 256 + tid;
  const float* xb = x + (size_t)b * CCH * NV + n;

  float qa[C8V], ka[C8V];
  #pragma unroll
  for (int o = 0; o < C8V; ++o) { qa[o] = bq[o]; ka[o] = bk[o]; }

  for (int c = 0; c < CCH; ++c) {
    const float xv = xb[(size_t)c * NV];
    #pragma unroll
    for (int o = 0; o < C8V; ++o) {
      qa[o] = fmaf(sWq[o * CCH + c], xv, qa[o]);
      ka[o] = fmaf(sWk[o * CCH + c], xv, ka[o]);
    }
  }

  union { _Float16 h[8]; v4u v; } pq, pk;
  #pragma unroll
  for (int o = 0; o < C8V; ++o) { pq.h[o] = (_Float16)qa[o]; pk.h[o] = (_Float16)ka[o]; }
  const size_t off = ((size_t)b * NV + n) * C8V;
  *(v4u*)(qT + off) = pq.v;
  *(v4u*)(kT + off) = pk.v;
}

// ---------------- Phase 1b: V projection (64 -> 64), f16 [b][c][n] ----------------
__global__ __launch_bounds__(256) void v_proj_kernel(
    const float* __restrict__ x,
    const float* __restrict__ Wv, const float* __restrict__ bv,
    _Float16* __restrict__ vT)
{
  __shared__ float sW[CCH];
  const int tid = threadIdx.x;
  const int blocksPerC = NV / 256;                       // 16
  const int b = blockIdx.x / (CCH * blocksPerC);
  const int c = (blockIdx.x / blocksPerC) % CCH;
  const int n = (blockIdx.x % blocksPerC) * 256 + tid;

  if (tid < CCH) sW[tid] = Wv[c * CCH + tid];
  __syncthreads();

  const float* xb = x + (size_t)b * CCH * NV + n;
  float acc = bv[c];
  for (int cc = 0; cc < CCH; ++cc) acc = fmaf(sW[cc], xb[(size_t)cc * NV], acc);
  vT[(size_t)b * CCH * NV + (size_t)c * NV + n] = (_Float16)acc;
}

// ---------------- Phase 2: fused flash attention ----------------
// One wave32 handles TWO 16-row i-tiles over one j-split of 2048.
// S^T tiles via v_wmma_f32_16x16x32_f16 (A = K rows, B = Q cols, K-dim 8 padded to 32).
// The f32 D-layout of S^T is exactly the f16 B-layout of the PV WMMA, so exp+pack
// is the only glue between the two GEMMs. K/V tile loads are shared by both i-tiles.
// Writes unnormalized (O, m, l) partials per split; combine_kernel merges splits.
__global__ __launch_bounds__(32) void attn_kernel(
    const _Float16* __restrict__ qT,
    const _Float16* __restrict__ kT,
    const _Float16* __restrict__ vT,
    float*          __restrict__ pO,    // [b][split][i][c] f32, unnormalized
    float*          __restrict__ pML)   // [b][split][i][{m,l}] f32
{
  const int lane = threadIdx.x;
  const int l16  = lane & 15;
  const int half = lane >> 4;

  const int grpPerB = NV / 32;                           // 128 i-groups of 32 rows
  const int b  = blockIdx.x / (NSPLIT * grpPerB);
  const int sp = (blockIdx.x / grpPerB) % NSPLIT;
  const int ig = blockIdx.x % grpPerB;
  const int jbeg = sp * JSPAN;
  const int jend = jbeg + JSPAN;

  const _Float16* qTb = qT + (size_t)b * NV * C8V;
  const _Float16* kTb = kT + (size_t)b * NV * C8V;
  const _Float16* vb  = vT + (size_t)b * CCH * NV;

  const v4u z4 = (v4u)0u;

  // B-operands: Q tiles, column i = l16, K(=c)=0..7 in half 0; zeros elsewhere
  V16H qB[2];
  #pragma unroll
  for (int it = 0; it < 2; ++it) {
    v4u d = *(const v4u*)(qTb + (size_t)(ig * 32 + it * 16 + l16) * C8V);
    qB[it].q[0] = half ? z4 : d;
    qB[it].q[1] = z4;
  }

  float m[2] = { -__builtin_inff(), -__builtin_inff() };
  float l[2] = { 0.0f, 0.0f };
  v8f Oacc[2][4] = {};                                    // O^T accumulators
  const v8f zeroC = {};

  for (int jc = jbeg; jc < jend; jc += 32) {
    // hint next K chunk toward the caches (emits global_prefetch)
    {
      const int jn = (jc + 32 < jend) ? (jc + 32) : jbeg;
      __builtin_prefetch(kTb + (size_t)(jn + l16) * C8V, 0, 1);
    }

    // --- K tiles (shared by both i-tiles), then 4 S^T tiles ---
    V16H kA[2];
    #pragma unroll
    for (int t = 0; t < 2; ++t) {
      v4u d = *(const v4u*)(kTb + (size_t)(jc + t * 16 + l16) * C8V);
      kA[t].q[0] = half ? z4 : d;
      kA[t].q[1] = z4;
    }
    v8f S[2][2];
    #pragma unroll
    for (int it = 0; it < 2; ++it) {
      #pragma unroll
      for (int t = 0; t < 2; ++t)
        S[it][t] = __builtin_amdgcn_wmma_f32_16x16x32_f16(
            false, kA[t].h, false, qB[it].h, (short)0, zeroC, false, false);
    }

    // --- online softmax: two independent chains (ILP hides exp/bpermute latency) ---
    V16H P[2];
    #pragma unroll
    for (int it = 0; it < 2; ++it) {
      float tmax = S[it][0][0];
      #pragma unroll
      for (int r = 1; r < 8; ++r) tmax = fmaxf(tmax, S[it][0][r]);
      #pragma unroll
      for (int r = 0; r < 8; ++r) tmax = fmaxf(tmax, S[it][1][r]);
      tmax = fmaxf(tmax, __shfl_xor(tmax, 16, 32));

      const float mnew = fmaxf(m[it], tmax);
      const float corr = __expf(m[it] - mnew);
      m[it] = mnew;
      l[it] *= corr;
      #pragma unroll
      for (int ct = 0; ct < 4; ++ct) {
        #pragma unroll
        for (int r = 0; r < 8; ++r) Oacc[it][ct][r] *= corr;
      }

      float p0[8], p1[8], ps = 0.0f;
      #pragma unroll
      for (int r = 0; r < 8; ++r) { p0[r] = __expf(S[it][0][r] - mnew); ps += p0[r]; }
      #pragma unroll
      for (int r = 0; r < 8; ++r) { p1[r] = __expf(S[it][1][r] - mnew); ps += p1[r]; }
      ps += __shfl_xor(ps, 16, 32);
      l[it] += ps;

      #pragma unroll
      for (int r = 0; r < 8; ++r) { P[it].h[r] = (_Float16)p0[r]; P[it].h[8 + r] = (_Float16)p1[r]; }
    }

    // --- O^T += V_tile x P^T : V tile loaded once, used by both i-tiles ---
    #pragma unroll
    for (int ct = 0; ct < 4; ++ct) {
      V16H vA;  // A-operand: row c = ct*16 + l16; K = jc + half*8 + {0..7} and +16
      const _Float16* vp = vb + (size_t)(ct * 16 + l16) * NV + jc + half * 8;
      vA.q[0] = *(const v4u*)(vp);
      vA.q[1] = *(const v4u*)(vp + 16);
      #pragma unroll
      for (int it = 0; it < 2; ++it)
        Oacc[it][ct] = __builtin_amdgcn_wmma_f32_16x16x32_f16(
            false, vA.h, false, P[it].h, (short)0, Oacc[it][ct], false, false);
    }
  }

  // --- write unnormalized partials for this split ---
  const size_t pbase = (size_t)(b * NSPLIT + sp) * NV;
  #pragma unroll
  for (int it = 0; it < 2; ++it) {
    const int i = ig * 32 + it * 16 + l16;
    float* po = pO + (pbase + i) * CCH + half * 8;
    #pragma unroll
    for (int ct = 0; ct < 4; ++ct) {
      v4f lo, hi;
      #pragma unroll
      for (int r = 0; r < 4; ++r) { lo[r] = Oacc[it][ct][r]; hi[r] = Oacc[it][ct][4 + r]; }
      *(v4f*)(po + ct * 16)     = lo;
      *(v4f*)(po + ct * 16 + 4) = hi;
    }
    if (half == 0) {
      pML[(pbase + i) * 2 + 0] = m[it];
      pML[(pbase + i) * 2 + 1] = l[it];
    }
  }
}

// ---------------- Phase 3: merge the two j-splits, normalize, add residual ----------------
__global__ __launch_bounds__(256) void combine_kernel(
    const float* __restrict__ x,
    const float* __restrict__ pO,
    const float* __restrict__ pML,
    float*       __restrict__ out)
{
  const int idx = blockIdx.x * 256 + threadIdx.x;        // b*NV + i, 8192 total
  const int b = idx / NV;
  const int i = idx % NV;

  const size_t r0 = (size_t)(b * NSPLIT + 0) * NV + i;
  const size_t r1 = (size_t)(b * NSPLIT + 1) * NV + i;
  const float m0 = pML[r0 * 2 + 0], l0 = pML[r0 * 2 + 1];
  const float m1 = pML[r1 * 2 + 0], l1 = pML[r1 * 2 + 1];
  const float mm = fmaxf(m0, m1);
  const float e0 = __expf(m0 - mm);
  const float e1 = __expf(m1 - mm);
  const float rl = 1.0f / (l0 * e0 + l1 * e1);

  const float* O0 = pO + r0 * CCH;
  const float* O1 = pO + r1 * CCH;
  const float* xb = x   + (size_t)b * CCH * NV + i;
  float*       ob = out + (size_t)b * CCH * NV + i;
  #pragma unroll 8
  for (int c = 0; c < CCH; ++c)
    ob[(size_t)c * NV] = (O0[c] * e0 + O1[c] * e1) * rl + xb[(size_t)c * NV];
}

// ---------------- host launcher ----------------
extern "C" void kernel_launch(void* const* d_in, const int* in_sizes, int n_in,
                              void* d_out, int out_size, void* d_ws, size_t ws_size,
                              hipStream_t stream) {
  const float* x  = (const float*)d_in[0];
  const float* Wq = (const float*)d_in[1];
  const float* bq = (const float*)d_in[2];
  const float* Wk = (const float*)d_in[3];
  const float* bk = (const float*)d_in[4];
  const float* Wv = (const float*)d_in[5];
  const float* bv = (const float*)d_in[6];
  float* out = (float*)d_out;

  _Float16* qT = (_Float16*)d_ws;                        // [B][N][8]          128 KB
  _Float16* kT = qT + (size_t)BATCH * NV * C8V;          // [B][N][8]          128 KB
  _Float16* vT = kT + (size_t)BATCH * NV * C8V;          // [B][C][N]            1 MB
  float*    pO  = (float*)(vT + (size_t)BATCH * CCH * NV);          // [B][2][N][C]  4 MB
  float*    pML = pO + (size_t)BATCH * NSPLIT * NV * CCH;           // [B][2][N][2] 128 KB

  qk_proj_kernel<<<BATCH * (NV / 256), 256, 0, stream>>>(x, Wq, bq, Wk, bk, qT, kT);
  v_proj_kernel<<<BATCH * CCH * (NV / 256), 256, 0, stream>>>(x, Wv, bv, vT);
  attn_kernel<<<BATCH * NSPLIT * (NV / 32), 32, 0, stream>>>(qT, kT, vT, pO, pML);
  combine_kernel<<<BATCH * NV / 256, 256, 0, stream>>>(x, pO, pML, out);
}
